// Decoder_13168369730058
// MI455X (gfx1250) — matrix operand order repaired and checked by reading, not compiled, not run on hardware
//
#include <hip/hip_runtime.h>
#include <hip/hip_fp16.h>

typedef __attribute__((ext_vector_type(16))) _Float16 v16h;
typedef __attribute__((ext_vector_type(8)))  _Float16 v8h;
typedef __attribute__((ext_vector_type(8)))  float    v8f;
typedef __attribute__((ext_vector_type(4)))  float    v4f;

#define BATCH 128
#define HDIM  512
#define ENCL  100

// ---- fast transcendentals: map straight to gfx1250 TRANS32 ops ----
__device__ __forceinline__ float fast_sigmoid(float x) {
    // 1/(1+e^-x) = rcp(1 + exp2(-x*log2e)) : v_exp_f32 + v_rcp_f32
    return __builtin_amdgcn_rcpf(1.0f + __builtin_amdgcn_exp2f(-x * 1.4426950408889634f));
}
__device__ __forceinline__ float fast_tanh(float x) {
#if __has_builtin(__builtin_amdgcn_tanhf)
    return __builtin_amdgcn_tanhf(x);           // v_tanh_f32 (CDNA5 TRANS32)
#else
    // tanh(x) = 1 - 2/(exp2(2x*log2e) + 1); saturates correctly at +/-1
    float e = __builtin_amdgcn_exp2f(x * 2.8853900817779268f);
    return 1.0f - 2.0f * __builtin_amdgcn_rcpf(e + 1.0f);
#endif
}

// --- extract h0 = encoder_outputs[0, :, -1, :] -> (128, 512) f32 ---
__global__ void gru_extract_h0(const float* __restrict__ enc, float* __restrict__ h0) {
    int i = blockIdx.x * 256 + threadIdx.x;          // 0 .. 128*512-1
    int b = i >> 9;
    int j = i & (HDIM - 1);
    h0[i] = enc[((size_t)b * ENCL + (ENCL - 1)) * HDIM + j];
}

// --- convert + fuse weights: Wc16[0:1536) = W_ih, Wc16[1536:3072) = W_hh (f16, row-major [3072][512]) ---
__global__ void gru_cvt_weights(const float* __restrict__ Wih, const float* __restrict__ Whh,
                                __half* __restrict__ Wc) {
    int i = blockIdx.x * 256 + threadIdx.x;          // 0 .. 1536*512-1
    _Float16* w = reinterpret_cast<_Float16*>(Wc);
    w[i]               = (_Float16)Wih[i];
    w[1536 * HDIM + i] = (_Float16)Whh[i];
}

// --- one GRU step: h_out = GRU(x, h_prev) with x == h_prev (or x == 0 when XZERO) ---
// grid = 32 blocks (16 output features each), 256 threads = 8 waves; wave w owns rows [16w, 16w+16).
// XZERO (t==0): i-gate GEMMs vanish (x==0), i-gates are pure bias.
template <bool XZERO>
__global__ __launch_bounds__(256, 1)
void gru_step(const __half* __restrict__ Wc_,    // fused [3072][512] f16
              const float* __restrict__ b_ih,    // [1536]
              const float* __restrict__ b_hh,    // [1536]
              const float* __restrict__ h_prev,  // [128][512] f32
              float* __restrict__ h_out) {       // [128][512] f32
    const _Float16* __restrict__ Wc = reinterpret_cast<const _Float16*>(Wc_);

    constexpr int G0 = XZERO ? 3 : 0;            // first active gate
    constexpr int NG = 6 - G0;                   // number of active gate GEMMs

    const int lane = threadIdx.x & 31;
    const int wave = threadIdx.x >> 5;
    const int m0   = wave * 16;                  // M tile base (batch rows)
    const int j0   = blockIdx.x * 16;            // output-feature tile base
    const int lj   = lane & 15;
    const int hi   = lane >> 4;                  // 0: lanes 0-15, 1: lanes 16-31
    const int j    = j0 + lj;

    // accumulators for active gates (order: g = G0 .. 5)
    v8f acc[NG];
#pragma unroll
    for (int g = 0; g < NG; ++g)
#pragma unroll
        for (int v = 0; v < 8; ++v) acc[g][v] = 0.0f;

    const int kbaseA = hi * 8;                   // A-fragment K sublane offset
    const int kloB   = hi * 8;                   // B-fragment K sublane offset
    const float* __restrict__ aRow = h_prev + (size_t)(m0 + lj) * HDIM;

    // per-gate weight row pointers: B column n = Wc row (g*512 + j), K contiguous
    const _Float16* wrow[NG];
#pragma unroll
    for (int g = 0; g < NG; ++g)
        wrow[g] = Wc + (size_t)((G0 + g) * HDIM + j) * HDIM + kloB;

    for (int kk = 0; kk < HDIM; kk += 32) {
        // ---- A fragment: rows m0..m0+15 of h_prev, K chunk [kk, kk+32), f32 -> f16 ----
        const float* ap = aRow + kk + kbaseA;
        v4f f0 = *(const v4f*)(ap);              // K = kb+0..3
        v4f f1 = *(const v4f*)(ap + 4);          // K = kb+4..7
        v4f f2 = *(const v4f*)(ap + 16);         // K = kb+16..19
        v4f f3 = *(const v4f*)(ap + 20);         // K = kb+20..23

        // ---- issue ALL B-fragment loads first (one clause), then convert A, then WMMAs ----
        v8h blo[NG], bhi[NG];
#pragma unroll
        for (int g = 0; g < NG; ++g) {
            const _Float16* wp = wrow[g] + kk;
            blo[g] = *(const v8h*)(wp);
            bhi[g] = *(const v8h*)(wp + 16);
        }

        v16h a;
#pragma unroll
        for (int i = 0; i < 4; ++i) {
            a[i]      = (_Float16)f0[i];
            a[4 + i]  = (_Float16)f1[i];
            a[8 + i]  = (_Float16)f2[i];
            a[12 + i] = (_Float16)f3[i];
        }

#pragma unroll
        for (int g = 0; g < NG; ++g) {
            v16h b;
#pragma unroll
            for (int i = 0; i < 8; ++i) { b[i] = blo[g][i]; b[8 + i] = bhi[g][i]; }
            acc[g] = __builtin_amdgcn_wmma_f32_16x16x32_f16(
                false, a, false, b, (short)0, acc[g], false, false);
        }
    }

    // ---- gating: D layout => this lane holds (M = m0 + v + 8*hi, N = j) for v = 0..7 ----
    const float bir = b_ih[j], biz = b_ih[HDIM + j], bin = b_ih[2 * HDIM + j];
    const float bhr = b_hh[j], bhz = b_hh[HDIM + j], bhn = b_hh[2 * HDIM + j];
#pragma unroll
    for (int v = 0; v < 8; ++v) {
        const int brow = m0 + v + hi * 8;
        const float hp = h_prev[(size_t)brow * HDIM + j];
        float ir, iz, in_, hr, hz, hn;
        if (XZERO) {
            ir = bir;  iz = biz;  in_ = bin;
            hr = acc[0][v] + bhr;  hz = acc[1][v] + bhz;  hn = acc[2][v] + bhn;
        } else {
            ir = acc[0][v] + bir;  iz = acc[1][v] + biz;  in_ = acc[2][v] + bin;
            hr = acc[3][v] + bhr;  hz = acc[4][v] + bhz;  hn = acc[5][v] + bhn;
        }
        const float r = fast_sigmoid(ir + hr);
        const float z = fast_sigmoid(iz + hz);
        const float n = fast_tanh(in_ + r * hn);
        h_out[(size_t)brow * HDIM + j] = (1.0f - z) * n + z * hp;
    }
}

extern "C" void kernel_launch(void* const* d_in, const int* in_sizes, int n_in,
                              void* d_out, int out_size, void* d_ws, size_t ws_size,
                              hipStream_t stream) {
    const float* enc  = (const float*)d_in[0];   // (1, 128, 100, 512)
    const float* Wih  = (const float*)d_in[1];   // (1536, 512)
    const float* Whh  = (const float*)d_in[2];   // (1536, 512)
    const float* bih  = (const float*)d_in[3];   // (1536,)
    const float* bhh  = (const float*)d_in[4];   // (1536,)
    float* out = (float*)d_out;                  // (out_len*128, 1, 512)

    const int stepElems = BATCH * HDIM;          // 65536
    const int out_len = out_size / stepElems;    // == d_in[5] value, recovered on host

    // workspace: [0, 256KB) h0 f32; [256KB, 256KB+3MB) fused f16 weights
    float*  h0  = (float*)d_ws;
    __half* Wc  = (__half*)((char*)d_ws + (size_t)stepElems * sizeof(float));

    gru_extract_h0<<<stepElems / 256, 256, 0, stream>>>(enc, h0);
    gru_cvt_weights<<<(1536 * HDIM) / 256, 256, 0, stream>>>(Wih, Whh, Wc);

    // t = 0: x == 0  -> 3-gate GEMM kernel
    gru_step<true><<<HDIM / 16, 256, 0, stream>>>(Wc, bih, bhh, h0, out);
    // t >= 1: x == h -> fused 6-gate GEMM kernel
    for (int t = 1; t < out_len; ++t) {
        gru_step<false><<<HDIM / 16, 256, 0, stream>>>(
            Wc, bih, bhh, out + (size_t)(t - 1) * stepElems, out + (size_t)t * stepElems);
    }
}
